// PointNet2Backbone_87402584474141
// MI455X (gfx1250) — matrix operand order repaired and checked by reading, not compiled, not run on hardware
//
#include <hip/hip_runtime.h>
#include <hip/hip_bf16.h>

typedef __attribute__((ext_vector_type(16))) __bf16 v16bf;
typedef __attribute__((ext_vector_type(8)))  __bf16 v8bf;
typedef __attribute__((ext_vector_type(8)))  float  v8f;

__device__ __forceinline__ __bf16 f2bf(float f) {
  unsigned u = __builtin_bit_cast(unsigned, f);
  unsigned r = u + 0x7FFFu + ((u >> 16) & 1u);   // round-to-nearest-even
  unsigned short h = (unsigned short)(r >> 16);
  return __builtin_bit_cast(__bf16, h);
}

__device__ __forceinline__ int imin(int a, int b) { return a < b ? a : b; }

// ---------------------------------------------------------------------------
// Pack f32 activations -> bf16, rows padded with zeros to CinP (mult. of 32).
// Removes all K guards from the GEMM and gives 16B-aligned A-fragment chunks.
// ---------------------------------------------------------------------------
__global__ void pack_x_bf16_kernel(const float* __restrict__ X, int Cin, int CinP,
                                   __bf16* __restrict__ Xp, long total) {
  long t = blockIdx.x * (long)blockDim.x + threadIdx.x;
  if (t >= total) return;
  const int col = (int)(t % CinP);
  const long row = t / CinP;
  const float v = (col < Cin) ? X[row * Cin + col] : 0.0f;
  Xp[t] = f2bf(v);
}

// ---------------------------------------------------------------------------
// Pack f32 weights -> bf16 in WMMA-B fragment order:
//   Wp[(tn*KB + kb)*256 + nn*16 + kk]  holds W[kb*16+kk, tn*16+nn]
// so each lane's 16 B values are 32 contiguous, 32B-aligned bytes.
// ---------------------------------------------------------------------------
__global__ void pack_w_bf16_kernel(const float* __restrict__ W, int Cin, int Cout,
                                   int KB, __bf16* __restrict__ Wp, long total) {
  long t = blockIdx.x * (long)blockDim.x + threadIdx.x;
  if (t >= total) return;
  const int kk = (int)(t % 16);
  const int nn = (int)((t >> 4) % 16);
  const long blk = t >> 8;
  const int kb = (int)(blk % KB);
  const int tn = (int)(blk / KB);
  const int k = kb * 16 + kk;
  const int n = tn * 16 + nn;
  const float v = (k < Cin && n < Cout) ? W[(long)k * Cout + n] : 0.0f;
  Wp[t] = f2bf(v);
}

// ---------------------------------------------------------------------------
// WMMA GEMM on pre-packed bf16: Y[R,Cout] = relu(X*W + bias), f32 accumulate.
// One wave32 per 16x32 output strip: per 32-wide K step the loop body is
// 6 x global_load_b128 + 2 x v_wmma_f32_16x16x32_bf16 (no conversions, no
// guards; OOB rows/cols clamped and simply not stored).
// ---------------------------------------------------------------------------
__global__ __launch_bounds__(128)
void gemm_bf16_wmma_kernel(const __bf16* __restrict__ Xp, int CinP,
                           const __bf16* __restrict__ Wp, int KB, int ntiles,
                           const float* __restrict__ bias,
                           float* __restrict__ Y,
                           int R, int Cout, int doRelu) {
  const int lane = threadIdx.x & 31;
  const int wave = threadIdx.x >> 5;
  const int tilesR  = (R + 15) >> 4;
  const int tilesC2 = (ntiles + 1) >> 1;
  long tile = (long)blockIdx.x * (blockDim.x >> 5) + wave;
  if (tile >= (long)tilesR * tilesC2) return;   // wave-uniform: EXEC all-1 for WMMA
  const int tr  = (int)(tile / tilesC2);
  const int tc2 = (int)(tile % tilesC2);
  const int m  = lane & 15;
  const int hi = lane >> 4;

  const int rowA  = tr * 16 + m;
  const int rowAc = imin(rowA, R - 1);          // clamp: OOB rows never stored
  const int tn0 = tc2 * 2;
  const int tn1 = imin(tn0 + 1, ntiles - 1);    // phantom tile clamped, not stored

  const __bf16* __restrict__ Xrow = Xp + (long)rowAc * CinP;
  const __bf16* __restrict__ Wb0  = Wp + ((long)tn0 * KB) * 256 + m * 16;
  const __bf16* __restrict__ Wb1  = Wp + ((long)tn1 * KB) * 256 + m * 16;

  v8f acc0 = {};
  v8f acc1 = {};
  for (int k0 = 0; k0 < CinP; k0 += 32) {
    const int kb = (k0 >> 4) + hi;
    const v8bf alo = *(const v8bf*)(Xrow + k0 + hi * 16);       // 16B aligned
    const v8bf ahi = *(const v8bf*)(Xrow + k0 + hi * 16 + 8);   // 16B aligned
    const v16bf a = __builtin_shufflevector(alo, ahi,
        0, 1, 2, 3, 4, 5, 6, 7, 8, 9, 10, 11, 12, 13, 14, 15);
    const v16bf b0 = *(const v16bf*)(Wb0 + (long)kb * 256);     // 32B aligned
    const v16bf b1 = *(const v16bf*)(Wb1 + (long)kb * 256);     // 32B aligned
    acc0 = __builtin_amdgcn_wmma_f32_16x16x32_bf16(false, a, false, b0,
                                                   (short)0, acc0, false, false);
    acc1 = __builtin_amdgcn_wmma_f32_16x16x32_bf16(false, a, false, b1,
                                                   (short)0, acc1, false, false);
  }

  const int n0 = tn0 * 16 + m;
  const int n1 = tn0 * 16 + 16 + m;
  const float bv0 = bias ? bias[imin(n0, Cout - 1)] : 0.0f;
  const float bv1 = bias ? bias[imin(n1, Cout - 1)] : 0.0f;
  #pragma unroll
  for (int v = 0; v < 8; ++v) {                 // C/D: VGPR v holds M = hi*8 + v
    const int row = tr * 16 + hi * 8 + v;
    if (row < R) {
      if (n0 < Cout) {
        float y = acc0[v] + bv0;
        if (doRelu) y = fmaxf(y, 0.0f);
        Y[(long)row * Cout + n0] = y;
      }
      if (n1 < Cout) {
        float y = acc1[v] + bv1;
        if (doRelu) y = fmaxf(y, 0.0f);
        Y[(long)row * Cout + n1] = y;
      }
    }
  }
}

// ---------------------------------------------------------------------------
// Farthest point sampling: one block per batch, iterative argmax reduction.
// ---------------------------------------------------------------------------
#define FPS_T 256
__global__ __launch_bounds__(FPS_T)
void fps_kernel(const float* __restrict__ pts, int stride, int N, int npoint,
                int* __restrict__ idx_out, float* __restrict__ dmin) {
  const int b = blockIdx.x;
  const float* p = pts + (long)b * N * stride;
  float* dm = dmin + (long)b * N;
  __shared__ float sval[FPS_T];
  __shared__ int   sidx[FPS_T];
  __shared__ int   s_far;
  for (int i = threadIdx.x; i < N; i += blockDim.x) dm[i] = 1e10f;
  if (threadIdx.x == 0) s_far = 0;
  __syncthreads();
  for (int it = 0; it < npoint; ++it) {
    const int far = s_far;
    if (threadIdx.x == 0) idx_out[b * npoint + it] = far;
    const float fx = p[(long)far * stride + 0];
    const float fy = p[(long)far * stride + 1];
    const float fz = p[(long)far * stride + 2];
    float best = -1.0f; int bi = 0;
    for (int i = threadIdx.x; i < N; i += blockDim.x) {
      const float dx = p[(long)i * stride + 0] - fx;
      const float dy = p[(long)i * stride + 1] - fy;
      const float dz = p[(long)i * stride + 2] - fz;
      float d = dx * dx + dy * dy + dz * dz;
      float dv = dm[i];
      if (d < dv) { dv = d; dm[i] = dv; }
      if (dv > best || (dv == best && i < bi)) { best = dv; bi = i; }
    }
    sval[threadIdx.x] = best; sidx[threadIdx.x] = bi;
    __syncthreads();
    for (int o = FPS_T / 2; o > 0; o >>= 1) {
      if (threadIdx.x < o) {
        const float ov = sval[threadIdx.x + o]; const int oi = sidx[threadIdx.x + o];
        if (ov > sval[threadIdx.x] || (ov == sval[threadIdx.x] && oi < sidx[threadIdx.x])) {
          sval[threadIdx.x] = ov; sidx[threadIdx.x] = oi;
        }
      }
      __syncthreads();
    }
    if (threadIdx.x == 0) s_far = sidx[0];
    __syncthreads();
  }
}

__global__ void gather3_kernel(const float* __restrict__ src, int stride, int N,
                               const int* __restrict__ idx, int S,
                               float* __restrict__ dst, long total) {
  long t = blockIdx.x * (long)blockDim.x + threadIdx.x;
  if (t >= total) return;
  const int c = (int)(t % 3);
  const long r = t / 3;          // b*S + s
  const int b = (int)(r / S);
  dst[t] = src[((long)b * N + idx[r]) * stride + c];
}

// Ball query: first nsample points (index order) within radius; pad with first hit (0 if none).
__global__ void ball_query_kernel(float r2, int nsample,
                                  const float* __restrict__ xyz, int xstride, int N,
                                  const float* __restrict__ new_xyz, int S,
                                  int* __restrict__ out, long total) {
  long t = blockIdx.x * (long)blockDim.x + threadIdx.x;
  if (t >= total) return;
  const int b = (int)(t / S);
  const float qx = new_xyz[t * 3 + 0];
  const float qy = new_xyz[t * 3 + 1];
  const float qz = new_xyz[t * 3 + 2];
  int* o = out + t * nsample;
  int cnt = 0, first = 0;
  for (int j = 0; j < N && cnt < nsample; ++j) {
    const float* pj = xyz + ((long)b * N + j) * xstride;
    const float dx = pj[0] - qx, dy = pj[1] - qy, dz = pj[2] - qz;
    if (dx * dx + dy * dy + dz * dz < r2) { if (cnt == 0) first = j; o[cnt++] = j; }
  }
  const int fill = (cnt > 0) ? first : 0;
  for (int k = cnt; k < nsample; ++k) o[k] = fill;
}

// H[r, 0:3] = xyz[gi]-center ; H[r, 3:C] = feats[gi]
__global__ void group_concat_kernel(const float* __restrict__ xyz, int xstride, int N,
                                    const float* __restrict__ feats, int fstride, int foff,
                                    const float* __restrict__ new_xyz,
                                    const int* __restrict__ gidx, int S, int nsample, int C,
                                    float* __restrict__ H, long total) {
  long t = blockIdx.x * (long)blockDim.x + threadIdx.x;
  if (t >= total) return;
  const int c = (int)(t % C);
  const long r = t / C;            // (b*S+s)*nsample + k
  const long sr = r / nsample;     // b*S + s
  const int b = (int)(sr / S);
  const long srow = (long)b * N + gidx[r];
  float v;
  if (c < 3) v = xyz[srow * xstride + c] - new_xyz[sr * 3 + c];
  else       v = feats[srow * fstride + foff + (c - 3)];
  H[t] = v;
}

__global__ void maxpool_kernel(const float* __restrict__ H, int nsample, int C,
                               float* __restrict__ out, int outStride, int colOff,
                               long total) {
  long t = blockIdx.x * (long)blockDim.x + threadIdx.x;
  if (t >= total) return;
  const int c = (int)(t % C);
  const long sr = t / C;
  const float* h = H + sr * (long)nsample * C + c;
  float m = h[0];
  for (int k = 1; k < nsample; ++k) m = fmaxf(m, h[(long)k * C]);
  out[sr * outStride + colOff + c] = m;
}

__global__ void bcast_concat_kernel(const float* __restrict__ lf, int C,
                                    float* __restrict__ out, int outStride, int colOff,
                                    int S, long total) {
  long t = blockIdx.x * (long)blockDim.x + threadIdx.x;
  if (t >= total) return;
  const int c = (int)(t % C);
  const long sr = t / C;
  const int b = (int)(sr / S);
  out[sr * outStride + colOff + c] = lf[(long)b * C + c];
}

__global__ void three_nn_kernel(const float* __restrict__ unknown, int ustride, int Nu,
                                const float* __restrict__ known, int Nk,
                                int* __restrict__ idx3, float* __restrict__ w3, long total) {
  long t = blockIdx.x * (long)blockDim.x + threadIdx.x;
  if (t >= total) return;
  const int b = (int)(t / Nu);
  const float* u = unknown + t * (long)ustride;
  const float ux = u[0], uy = u[1], uz = u[2];
  float d0 = 1e30f, d1 = 1e30f, d2 = 1e30f;
  int i0 = 0, i1 = 0, i2 = 0;
  for (int j = 0; j < Nk; ++j) {
    const float* p = known + ((long)b * Nk + j) * 3;
    const float dx = p[0] - ux, dy = p[1] - uy, dz = p[2] - uz;
    const float d = dx * dx + dy * dy + dz * dz;
    if (d < d0)      { d2 = d1; i2 = i1; d1 = d0; i1 = i0; d0 = d; i0 = j; }
    else if (d < d1) { d2 = d1; i2 = i1; d1 = d; i1 = j; }
    else if (d < d2) { d2 = d; i2 = j; }
  }
  const float w0 = 1.0f / (fmaxf(d0, 0.0f) + 1e-8f);
  const float w1 = 1.0f / (fmaxf(d1, 0.0f) + 1e-8f);
  const float w2 = 1.0f / (fmaxf(d2, 0.0f) + 1e-8f);
  const float s = w0 + w1 + w2;
  idx3[t * 3 + 0] = i0; idx3[t * 3 + 1] = i1; idx3[t * 3 + 2] = i2;
  w3[t * 3 + 0] = w0 / s; w3[t * 3 + 1] = w1 / s; w3[t * 3 + 2] = w2 / s;
}

__global__ void interp_concat_kernel(const float* __restrict__ kf, int Ck, int Nk,
                                     const float* __restrict__ uf, int ufs, int ufo, int Cu,
                                     const int* __restrict__ idx3, const float* __restrict__ w3,
                                     int Nu, float* __restrict__ H, long total) {
  const int C = Ck + Cu;
  long t = blockIdx.x * (long)blockDim.x + threadIdx.x;
  if (t >= total) return;
  const int c = (int)(t % C);
  const long r = t / C;            // b*Nu + n
  const int b = (int)(r / Nu);
  if (c < Ck) {
    const int* ii = idx3 + r * 3;
    const float* ww = w3 + r * 3;
    float acc = 0.0f;
    for (int k = 0; k < 3; ++k)
      acc += ww[k] * kf[((long)b * Nk + ii[k]) * Ck + c];
    H[t] = acc;
  } else {
    H[t] = uf[r * (long)ufs + ufo + (c - Ck)];
  }
}

// src (B,S,C) -> dst (B,C,S)
__global__ void transpose_out_kernel(const float* __restrict__ src, int S, int C,
                                     float* __restrict__ dst, long total) {
  long t = blockIdx.x * (long)blockDim.x + threadIdx.x;
  if (t >= total) return;
  const int c = (int)(t % C);
  const long r = t / C;
  const int s = (int)(r % S);
  const int b = (int)(r / S);
  dst[((long)b * C + c) * S + s] = src[t];
}

// ---------------------------------------------------------------------------
// Host orchestration
// ---------------------------------------------------------------------------
extern "C" void kernel_launch(void* const* d_in, const int* in_sizes, int n_in,
                              void* d_out, int out_size, void* d_ws, size_t ws_size,
                              hipStream_t stream) {
  (void)in_sizes; (void)n_in; (void)out_size; (void)ws_size;
  const int B = 4, N0 = 8192;

  static const int   sa_npoint[4]     = {1024, 256, 64, 16};
  static const float sa_radius[4][2]  = {{0.05f,0.1f},{0.1f,0.2f},{0.2f,0.4f},{0.4f,0.8f}};
  static const int   sa_ns[2]         = {16, 32};
  static const int   sa_dims[4][2][4] = {
    {{6,16,16,32},     {6,32,32,64}},
    {{99,64,64,128},   {99,64,96,128}},
    {{515,128,196,256},{515,128,196,256}},
    {{515,256,256,512},{515,256,384,512}},
  };
  static const int fp_dims[4][3] = {{1536,512,512},{1024,512,512},{608,256,256},{259,128,128}};

  const float* pc  = (const float*)d_in[0];
  const float* lfe = (const float*)d_in[1];
  const float* saW[4][2][3]; const float* saB[4][2][3];
  const float* fpW[4][2];    const float* fpB[4][2];
  int pi = 2;
  for (int si = 0; si < 4; ++si)
    for (int sc = 0; sc < 2; ++sc)
      for (int l = 0; l < 3; ++l) {
        saW[si][sc][l] = (const float*)d_in[pi++];
        saB[si][sc][l] = (const float*)d_in[pi++];
      }
  for (int fi = 0; fi < 4; ++fi)
    for (int l = 0; l < 2; ++l) {
      fpW[fi][l] = (const float*)d_in[pi++];
      fpB[fi][l] = (const float*)d_in[pi++];
    }

  // ---- workspace bump allocator (256B-aligned slices) ----
  float* ws = (float*)d_ws;
  size_t off = 0;
  auto alloc = [&](size_t n) -> float* {
    float* p = ws + off; off += (n + 63) & ~(size_t)63; return p;
  };
  float* dmin   = alloc((size_t)B * N0);
  int*   fpsIdx = (int*)alloc((size_t)B * 1024);
  int*   gidx   = (int*)alloc((size_t)B * 1024 * 32);
  int*   idx3   = (int*)alloc((size_t)B * N0 * 3);
  float* w3     = alloc((size_t)B * N0 * 3);
  float* xyz1 = alloc((size_t)B * 1024 * 3);
  float* xyz2 = alloc((size_t)B * 256 * 3);
  float* xyz3 = alloc((size_t)B * 64 * 3);
  float* xyz4 = alloc((size_t)B * 16 * 3);
  float* f1 = alloc((size_t)B * 1024 * 96);
  float* f2 = alloc((size_t)B * 256 * 512);
  float* f3 = alloc((size_t)B * 64 * 512);
  float* f4 = alloc((size_t)B * 16 * 1024);
  float* lf1 = alloc((size_t)B * 64 * 512);
  float* lf2 = alloc((size_t)B * 256 * 512);
  float* lf3 = alloc((size_t)B * 1024 * 256);
  const size_t BUFN = 8519680;                   // >= B*8192*259 (largest activation)
  float* bufA = alloc(BUFN);
  float* bufB = alloc(BUFN);
  __bf16* xpBuf = (__bf16*)alloc(4755456);       // 9,510,912 bf16 >= 32768*290 rows*CinP
  __bf16* wpBuf = (__bf16*)alloc(400000);        // 800,000 bf16 >= 32*96*256 (fp1 W)

  auto grid = [](long total) { return dim3((unsigned)((total + 255) / 256)); };

  auto gemm = [&](const float* X, const float* Wm, const float* bs, float* Y,
                  int R, int Cin, int Cout) {
    const int CinP = (Cin + 31) & ~31;
    const int KB = CinP >> 4;
    const int ntiles = (Cout + 15) >> 4;
    long tx = (long)R * CinP;
    pack_x_bf16_kernel<<<grid(tx), 256, 0, stream>>>(X, Cin, CinP, xpBuf, tx);
    long tw = (long)ntiles * KB * 256;
    pack_w_bf16_kernel<<<grid(tw), 256, 0, stream>>>(Wm, Cin, Cout, KB, wpBuf, tw);
    long tiles = (long)((R + 15) / 16) * ((ntiles + 1) / 2);
    unsigned blocks = (unsigned)((tiles + 3) / 4);           // 4 waves / block
    gemm_bf16_wmma_kernel<<<blocks, 128, 0, stream>>>(xpBuf, CinP, wpBuf, KB, ntiles,
                                                      bs, Y, R, Cout, 1);
  };

  auto run_sa = [&](int si, const float* inXyz, int xs, int N,
                    const float* inF, int fs, int fo, int F,
                    float* outXyz, float* outF, int outStride) {
    const int S = sa_npoint[si];
    fps_kernel<<<B, FPS_T, 0, stream>>>(inXyz, xs, N, S, fpsIdx, dmin);
    long tg = (long)B * S * 3;
    gather3_kernel<<<grid(tg), 256, 0, stream>>>(inXyz, xs, N, fpsIdx, S, outXyz, tg);
    int colOff = 0;
    for (int sc = 0; sc < 2; ++sc) {
      const float r = sa_radius[si][sc];
      const int   ns = sa_ns[sc];
      long tq = (long)B * S;
      ball_query_kernel<<<grid(tq), 256, 0, stream>>>(r * r, ns, inXyz, xs, N,
                                                      outXyz, S, gidx, tq);
      const int* dims = sa_dims[si][sc];
      const int C0 = 3 + F;
      long th = (long)B * S * ns * C0;
      group_concat_kernel<<<grid(th), 256, 0, stream>>>(inXyz, xs, N, inF, fs, fo,
                                                        outXyz, gidx, S, ns, C0, bufA, th);
      const int R = B * S * ns;
      gemm(bufA, saW[si][sc][0], saB[si][sc][0], bufB, R, dims[0], dims[1]);
      gemm(bufB, saW[si][sc][1], saB[si][sc][1], bufA, R, dims[1], dims[2]);
      gemm(bufA, saW[si][sc][2], saB[si][sc][2], bufB, R, dims[2], dims[3]);
      long tp = (long)B * S * dims[3];
      maxpool_kernel<<<grid(tp), 256, 0, stream>>>(bufB, ns, dims[3], outF,
                                                   outStride, colOff, tp);
      colOff += dims[3];
    }
  };

  auto run_fp = [&](int fi, const float* uXyz, int us, int Nu,
                    const float* kXyz, int Nk,
                    const float* uF, int ufs, int ufo, int Cu,
                    const float* kF, int Ck, float* outLf) {
    long tn = (long)B * Nu;
    three_nn_kernel<<<grid(tn), 256, 0, stream>>>(uXyz, us, Nu, kXyz, Nk, idx3, w3, tn);
    const int* dims = fp_dims[fi];
    long th = tn * dims[0];
    interp_concat_kernel<<<grid(th), 256, 0, stream>>>(kF, Ck, Nk, uF, ufs, ufo, Cu,
                                                       idx3, w3, Nu, bufA, th);
    gemm(bufA, fpW[fi][0], fpB[fi][0], bufB, (int)tn, dims[0], dims[1]);
    gemm(bufB, fpW[fi][1], fpB[fi][1], bufA, (int)tn, dims[1], dims[2]);
    if (outLf)
      hipMemcpyAsync(outLf, bufA, (size_t)tn * dims[2] * sizeof(float),
                     hipMemcpyDeviceToDevice, stream);
  };

  // ---- encoder (SA) ----
  run_sa(0, pc,   6, N0,   pc, 6, 3, 3,   xyz1, f1, 96);
  run_sa(1, xyz1, 3, 1024, f1, 96, 0, 96, xyz2, f2, 512);
  long te = (long)B * 256 * 256;
  bcast_concat_kernel<<<grid(te), 256, 0, stream>>>(lfe, 256, f2, 512, 256, 256, te);
  run_sa(2, xyz2, 3, 256,  f2, 512, 0, 512, xyz3, f3, 512);
  run_sa(3, xyz3, 3, 64,   f3, 512, 0, 512, xyz4, f4, 1024);

  // ---- decoder (FP) ----
  run_fp(0, xyz3, 3, 64,   xyz4, 16,   f3, 512, 0, 512, f4,  1024, lf1);
  run_fp(1, xyz2, 3, 256,  xyz3, 64,   f2, 512, 0, 512, lf1, 512,  lf2);
  run_fp(2, xyz1, 3, 1024, xyz2, 256,  f1, 96,  0, 96,  lf2, 512,  lf3);
  run_fp(3, pc,   6, N0,   xyz1, 1024, pc, 6,   3, 3,   lf3, 256,  nullptr); // result in bufA

  // ---- outputs: (sa4_f.T, lf.T) ----
  float* out = (float*)d_out;
  long t0 = (long)B * 16 * 1024;
  transpose_out_kernel<<<grid(t0), 256, 0, stream>>>(f4, 16, 1024, out, t0);
  long t1 = (long)B * N0 * 128;
  transpose_out_kernel<<<grid(t1), 256, 0, stream>>>(bufA, N0, 128, out + t0, t1);
}
